// MAM_68418829025563
// MI455X (gfx1250) — matrix operand order repaired and checked by reading, not compiled, not run on hardware
//
#include <hip/hip_runtime.h>
#include <math.h>

// ---------------------------------------------------------------------------
// MAM fused retrieval-attention for MI455X (gfx1250, wave32).
//   query = x @ w1^T + b1
//   q2    = query @ w2            (b2 and query·b2 cancel: softmax shift-inv)
//   scores[b,k] = keys[b,k,:]·q2[b,:] / sqrt(D)
//   out = 0.5*x + 0.5*softmax(scores) @ values
// Compute collapses to ~9 GFLOP; 1.1 GB streamed -> HBM-bound (~47us floor).
// ---------------------------------------------------------------------------

typedef __attribute__((ext_vector_type(16))) __bf16 v16bf;
typedef __attribute__((ext_vector_type(8)))  __bf16 v8bf;
typedef __attribute__((ext_vector_type(4)))  __bf16 v4bf;
typedef __attribute__((ext_vector_type(8)))  float  v8f;
typedef __attribute__((ext_vector_type(4)))  float  f32x4;
typedef __attribute__((ext_vector_type(2)))  float  f32x2;
typedef __attribute__((ext_vector_type(4)))  unsigned int u32x4;
typedef __attribute__((ext_vector_type(8)))  int          i32x8;
typedef __attribute__((ext_vector_type(4)))  int          i32x4;

constexpr int   BATCH = 8192;
constexpr int   KNBR  = 32;
constexpr int   DIM   = 512;
constexpr float SCALE = 0.04419417382415922f;   // 1/sqrt(512)

constexpr int BM = 64, BN = 64, BK = 32;
constexpr int LDP = 40;   // padded LDS row stride in bf16 (80B, 16B-aligned)

#if defined(__AMDGCN__) && __has_builtin(__builtin_amdgcn_tensor_load_to_lds)
#define MAM_HAVE_TDM 1
#if __has_include(<hip/amd_detail/amd_gfx1250_TDM.h>)
#define MAM_TDM_6ARG 1
#endif
#endif

__device__ __forceinline__ void cvt4(const f32x4 f, v4bf& hi, v4bf& lo) {
#pragma unroll
  for (int j = 0; j < 4; ++j) {
    const __bf16 h = (__bf16)f[j];
    hi[j] = h;
    lo[j] = (__bf16)(f[j] - (float)h);
  }
}

__device__ __forceinline__ v16bf frag2(const __bf16* p0, const __bf16* p1) {
  const v8bf a = *(const v8bf*)p0;
  const v8bf b = *(const v8bf*)p1;
  return __builtin_shufflevector(a, b, 0, 1, 2, 3, 4, 5, 6, 7,
                                 8, 9, 10, 11, 12, 13, 14, 15);
}

// ---------------------------------------------------------------------------
// Split-precision bf16 WMMA GEMM:  out[m,n] = sum_k A[m,k] * W(k,n)  (+bias[n])
//   TRANS_B=true : W(k,n) = W[n*DIM + k]   (w1; staging is a straight copy)
//   TRANS_B=false: W(k,n) = W[k*DIM + n]   (w2; transposed into LDS)
// f32 operands decomposed hi+lo bf16; acc += aH*bH + aH*bL + aL*bH gives
// ~2^-16 relative error (f32-quality) on the bf16 WMMA pipe.
// LDS tiles are fragment-major:
//   A [row][k]  : lane fragment = 2 contiguous 8-elem runs -> 2x ds_load_b128
//   B [col][k]  : lane fragment = 2 contiguous 8-elem runs -> 2x ds_load_b128
// ---------------------------------------------------------------------------
template <bool TRANS_B, bool HAS_BIAS>
__launch_bounds__(256)
__global__ void mam_gemm_bf16x2(const float* __restrict__ A,
                                const float* __restrict__ W,
                                const float* __restrict__ bias,
                                float* __restrict__ out) {
  __shared__ __align__(16) __bf16 aHi[BM][LDP];
  __shared__ __align__(16) __bf16 aLo[BM][LDP];
  __shared__ __align__(16) __bf16 bHi[BN][LDP];   // [col][k]
  __shared__ __align__(16) __bf16 bLo[BN][LDP];

  const int tid  = threadIdx.x;
  const int lane = tid & 31;
  const int wave = tid >> 5;
  const int half = lane >> 4;    // 0: lanes 0-15, 1: lanes 16-31
  const int l16  = lane & 15;

  const int m0 = blockIdx.x * BM;
  const int n0 = blockIdx.y * BN;

  const int mt  = wave >> 1;        // wave's m-tile (0..3)
  const int nt0 = (wave & 1) * 2;   // wave's first n-tile (0 or 2)

  v8f acc0 = {};
  v8f acc1 = {};

  for (int kk = 0; kk < DIM; kk += BK) {
    // ---- stage A [BM x BK] as hi/lo bf16, vectorized f32x4 -> v4bf ----
#pragma unroll
    for (int q = tid; q < BM * BK / 4; q += 256) {
      const int r = q >> 3, c4 = (q & 7) * 4;
      const f32x4 f = *(const f32x4*)(A + (size_t)(m0 + r) * DIM + kk + c4);
      v4bf h, l;
      cvt4(f, h, l);
      *(v4bf*)&aHi[r][c4] = h;
      *(v4bf*)&aLo[r][c4] = l;
    }
    // ---- stage B [BN cols x BK k] (col-major-in-LDS) ----
    if (TRANS_B) {
      // global w1 is [n][k]: direct vectorized copy
#pragma unroll
      for (int q = tid; q < BN * BK / 4; q += 256) {
        const int n = q >> 3, c4 = (q & 7) * 4;
        const f32x4 f = *(const f32x4*)(W + (size_t)(n0 + n) * DIM + kk + c4);
        v4bf h, l;
        cvt4(f, h, l);
        *(v4bf*)&bHi[n][c4] = h;
        *(v4bf*)&bLo[n][c4] = l;
      }
    } else {
      // global w2 is [k][n]: coalesced f32x4 read, scalar transpose stores
#pragma unroll
      for (int q = tid; q < BN * BK / 4; q += 256) {
        const int k = q >> 4, n4 = (q & 15) * 4;
        const f32x4 f = *(const f32x4*)(W + (size_t)(kk + k) * DIM + n0 + n4);
        v4bf h, l;
        cvt4(f, h, l);
#pragma unroll
        for (int j = 0; j < 4; ++j) {
          bHi[n4 + j][k] = h[j];
          bLo[n4 + j][k] = l[j];
        }
      }
    }
    __syncthreads();

    // ---- fragments via wide LDS loads (ISA 7.12.2 layouts) ----
    const __bf16* arH = &aHi[mt * 16 + l16][0];
    const __bf16* arL = &aLo[mt * 16 + l16][0];
    const v16bf afH = frag2(arH + 8 * half, arH + 16 + 8 * half);
    const v16bf afL = frag2(arL + 8 * half, arL + 16 + 8 * half);

#pragma unroll
    for (int t = 0; t < 2; ++t) {
      const __bf16* bcH = &bHi[(nt0 + t) * 16 + l16][0];
      const __bf16* bcL = &bLo[(nt0 + t) * 16 + l16][0];
      const v16bf bfH = frag2(bcH + 8 * half, bcH + 16 + 8 * half);
      const v16bf bfL = frag2(bcL + 8 * half, bcL + 16 + 8 * half);

      v8f& acc = t ? acc1 : acc0;
      acc = __builtin_amdgcn_wmma_f32_16x16x32_bf16(false, afH, false, bfH,
                                                    (short)0, acc, false, false);
      acc = __builtin_amdgcn_wmma_f32_16x16x32_bf16(false, afH, false, bfL,
                                                    (short)0, acc, false, false);
      acc = __builtin_amdgcn_wmma_f32_16x16x32_bf16(false, afL, false, bfH,
                                                    (short)0, acc, false, false);
    }
    __syncthreads();
  }

  // ---- epilogue: C/D layout VGPR v -> M = v + 8*half, N = l16 ----
#pragma unroll
  for (int t = 0; t < 2; ++t) {
    const v8f& acc = t ? acc1 : acc0;
    const int col = n0 + (nt0 + t) * 16 + l16;
    float bv = 0.0f;
    if (HAS_BIAS) bv = bias[col];
#pragma unroll
    for (int v = 0; v < 8; ++v) {
      const int row = m0 + mt * 16 + half * 8 + v;
      out[(size_t)row * DIM + col] = acc[v] + bv;
    }
  }
}

// ---------------------------------------------------------------------------
// Streaming attention: one workgroup (256 thr = 8 wave32) per batch row.
// q2[b] row (2KB) is brought into LDS by the Tensor Data Mover (1-D tile);
// keys/values streamed once with nontemporal loads (bypass-friendly, the
// 1GB stream exceeds the 192MB L2 anyway).
// ---------------------------------------------------------------------------
__launch_bounds__(256)
__global__ void mam_attention(const float* __restrict__ x,
                              const float* __restrict__ keys,
                              const float* __restrict__ values,
                              const float* __restrict__ q2,
                              float* __restrict__ out) {
  __shared__ __align__(16) float q2s[DIM];
  __shared__ float scores[KNBR];
  __shared__ float attn[KNBR];

  const int b    = blockIdx.x;
  const int tid  = threadIdx.x;
  const int lane = tid & 31;
  const int wave = tid >> 5;

  // 1) q2[b] -> LDS
#if defined(MAM_HAVE_TDM)
  if (wave == 0) {
    const unsigned long long ga = (unsigned long long)(const void*)(q2 + (size_t)b * DIM);
    const unsigned lds = (unsigned)(unsigned long long)(void*)&q2s[0];
    // D# group0: count=1, lds_addr, global_addr[56:0], type=2
    u32x4 g0;
    g0[0] = 1u;                                   // count=1, no gather
    g0[1] = lds;                                  // lds_addr
    g0[2] = (unsigned)(ga & 0xFFFFFFFFu);         // global_addr[31:0]
    g0[3] = (unsigned)((ga >> 32) & 0x01FFFFFFu) | (2u << 30);  // [56:32] | type=2
    // D# group1: data_size=4B (code 2), tensor_dim0=512, tile_dim0=512, 1-D
    i32x8 g1 = {};
    g1[0] = 2 << 16;                              // data_size = 4 bytes
    g1[1] = DIM << 16;                            // tensor_dim0[15:0] in bits 63:48
    g1[2] = 0;                                    // tensor_dim0 hi, tensor_dim1 lo
    g1[3] = DIM << 16;                            // tile_dim0 in bits 127:112
    g1[4] = 0;                                    // tile_dim1/2 unused
    g1[5] = DIM;                                  // tensor_dim0_stride low
    g1[6] = 0;
    g1[7] = 0;
    const i32x4 gz = {};
#if defined(MAM_TDM_6ARG)
    const i32x8 gz8 = {};
    __builtin_amdgcn_tensor_load_to_lds(g0, g1, gz, gz, gz8, 0);
#else
    __builtin_amdgcn_tensor_load_to_lds(g0, g1, gz, gz, 0);
#endif
    __builtin_amdgcn_s_wait_tensorcnt(0);
  }
#else
  {
    const f32x2 v = *(const f32x2*)(q2 + (size_t)b * DIM + tid * 2);
    *(f32x2*)&q2s[tid * 2] = v;
  }
#endif
  __syncthreads();

  // 2) scores: each wave handles 4 neighbors; per-lane f32x4 partial dots
  const float* kb = keys + ((size_t)b * KNBR + wave * 4) * DIM;
#pragma unroll
  for (int j = 0; j < 4; ++j) {
    const float* kr = kb + j * DIM;
    if (j < 3) __builtin_prefetch(kb + (j + 1) * DIM + lane * 4, 0, 0);
    float acc = 0.0f;
#pragma unroll
    for (int d = 0; d < DIM; d += 128) {
      const f32x4 kv = __builtin_nontemporal_load((const f32x4*)(kr + d + lane * 4));
      const f32x4 qv = *(const f32x4*)&q2s[d + lane * 4];
      acc += kv[0] * qv[0] + kv[1] * qv[1] + kv[2] * qv[2] + kv[3] * qv[3];
    }
#pragma unroll
    for (int m = 16; m >= 1; m >>= 1) acc += __shfl_xor(acc, m, 32);
    if (lane == 0) scores[wave * 4 + j] = acc * SCALE;
  }
  __syncthreads();

  // 3) softmax over K=32 (query·b2 shift cancels; omitted)
  if (tid < KNBR) {
    float mx = -1e30f;
#pragma unroll
    for (int k = 0; k < KNBR; ++k) mx = fmaxf(mx, scores[k]);
    float s = 0.0f;
#pragma unroll
    for (int k = 0; k < KNBR; ++k) s += __expf(scores[k] - mx);
    attn[tid] = __expf(scores[tid] - mx) / s;
  }
  __syncthreads();

  // 4) combined = attn @ values, fused residual blend; 2 floats per thread
  const int d0 = tid * 2;
  const float* vb = values + (size_t)b * KNBR * DIM + d0;
  f32x2 accv = {0.0f, 0.0f};
#pragma unroll
  for (int k = 0; k < KNBR; ++k) {
    const float w = attn[k];
    const f32x2 vv = __builtin_nontemporal_load((const f32x2*)(vb + k * DIM));
    accv[0] += w * vv[0];
    accv[1] += w * vv[1];
  }
  const f32x2 xv = *(const f32x2*)(x + (size_t)b * DIM + d0);
  f32x2 o;
  o[0] = 0.5f * xv[0] + 0.5f * accv[0];
  o[1] = 0.5f * xv[1] + 0.5f * accv[1];
  *(f32x2*)(out + (size_t)b * DIM + d0) = o;
}

// ---------------------------------------------------------------------------
extern "C" void kernel_launch(void* const* d_in, const int* in_sizes, int n_in,
                              void* d_out, int out_size, void* d_ws, size_t ws_size,
                              hipStream_t stream) {
  (void)in_sizes; (void)n_in; (void)out_size; (void)ws_size;
  const float* x    = (const float*)d_in[0];
  const float* keys = (const float*)d_in[1];
  const float* vals = (const float*)d_in[2];
  const float* w1   = (const float*)d_in[3];
  const float* b1   = (const float*)d_in[4];
  const float* w2   = (const float*)d_in[5];
  // d_in[6] (b2) is mathematically irrelevant: softmax is shift-invariant.
  float* out   = (float*)d_out;

  float* query = (float*)d_ws;                   // [BATCH, DIM]
  float* q2    = query + (size_t)BATCH * DIM;    // [BATCH, DIM]

  const dim3 grid(BATCH / BM, DIM / BN);
  mam_gemm_bf16x2<true,  true ><<<grid, 256, 0, stream>>>(x,     w1, b1,      query);
  mam_gemm_bf16x2<false, false><<<grid, 256, 0, stream>>>(query, w2, nullptr, q2);
  mam_attention<<<BATCH, 256, 0, stream>>>(x, keys, vals, q2, out);
}
